// DIFFormer_14594298872390
// MI455X (gfx1250) — matrix operand order repaired and checked by reading, not compiled, not run on hardware
//
#include <hip/hip_runtime.h>

#define NN      100000
#define EE      800000
#define IN_DIM  256
#define HID     64
#define NLAYERS 2
#define ALPHA   0.5f
#define LN_EPS  1e-5f

typedef __attribute__((ext_vector_type(16))) __bf16 v16bf;
typedef __attribute__((ext_vector_type(8)))  __bf16 v8bf;
typedef __attribute__((ext_vector_type(4)))  __bf16 v4bf;
typedef __attribute__((ext_vector_type(8)))  float  v8f;
typedef __attribute__((ext_vector_type(4)))  float  v4f;

// ---------------------------------------------------------------------------
// WMMA helper: D = A(16x32 bf16) * B(32x16 bf16) + C(16x16 f32)
// ---------------------------------------------------------------------------
static __device__ __forceinline__ v8f wmma_bf16(v16bf a, v16bf b, v8f c) {
  return __builtin_amdgcn_wmma_f32_16x16x32_bf16(
      /*neg_a=*/false, a, /*neg_b=*/false, b,
      /*c_mod=*/(short)0, c, /*reuse_a=*/false, /*reuse_b=*/false);
}

// A fragment (16x32 MxK) from row-major bf16 [.., lda].
// lane L(0..15): M=L, K = 0..7 then 16..23; lane L+16: K = 8..15 then 24..31.
// Per lane: two contiguous 8-element (16B) runs -> vector loads.
static __device__ __forceinline__ v16bf load_a_bf(const __bf16* __restrict__ A,
                                                  int lda, int row0, int k0,
                                                  int lane) {
  const __bf16* base =
      A + (size_t)(row0 + (lane & 15)) * lda + k0 + ((lane >> 4) << 3);
  v8bf lo = *(const v8bf*)base;
  v8bf hi = *(const v8bf*)(base + 16);
  v16bf r;
#pragma unroll
  for (int i = 0; i < 8; ++i) { r[i] = lo[i]; r[8 + i] = hi[i]; }
  return r;
}

// Packed-B fragment: P laid out as [frag][lane][16] contiguous bf16 (32B/lane).
static __device__ __forceinline__ v16bf load_b_packed(const __bf16* __restrict__ P,
                                                      int frag, int lane) {
  return *(const v16bf*)(P + (((size_t)frag << 5) + lane) * 16);
}

// ---------------------------------------------------------------------------
// Utility kernels
// ---------------------------------------------------------------------------
__global__ void k_zero(float* p, int n) {
  int t = blockIdx.x * blockDim.x + threadIdx.x;
  if (t < n) p[t] = 0.0f;
}

// Pack row-major fp32 W[K][ncol_src] into per-lane WMMA B fragments (bf16):
// dst[((ks*ntiles + j)*32 + lane)*16 + i] = W[ks*32 + ((lane>>4)<<3) + (i&7) + ((i>>3)<<4)][j*16 + (lane&15)]
// Columns >= ncol_src are zero-padded (ncol_dst may exceed ncol_src).
__global__ void k_packB(const float* __restrict__ src, __bf16* __restrict__ dst,
                        int K, int ncol_dst, int ncol_src) {
  int t = blockIdx.x * blockDim.x + threadIdx.x;
  int ntiles = ncol_dst >> 4;
  int total = (K >> 5) * ntiles * 512;
  if (t >= total) return;
  int i    = t & 15;
  int lane = (t >> 4) & 31;
  int frag = t >> 9;
  int j  = frag % ntiles;
  int ks = frag / ntiles;
  int k = ks * 32 + ((lane >> 4) << 3) + (i & 7) + ((i >> 3) << 4);
  int c = j * 16 + (lane & 15);
  dst[t] = (c < ncol_src) ? (__bf16)src[(size_t)k * ncol_src + c] : (__bf16)0.0f;
}

__global__ void k_deg(const int* __restrict__ col, float* __restrict__ deg) {
  int t = blockIdx.x * blockDim.x + threadIdx.x;
  if (t < EE) unsafeAtomicAdd(&deg[col[t]], 1.0f);
}

// ---------------------------------------------------------------------------
// h = relu(LN(x @ fc0_w + fc0_b));  writes fp32 h and bf16 copy hb.
// blockDim=64 (2 waves), each wave owns a 16-row tile, 4 col tiles of 16.
// ---------------------------------------------------------------------------
__global__ void k_fc0(const float* __restrict__ x, const __bf16* __restrict__ Wp,
                      const float* __restrict__ bias,
                      const float* __restrict__ lnw, const float* __restrict__ lnb,
                      float* __restrict__ h, __bf16* __restrict__ hb) {
  __shared__ float tile[2][16][72];
  int wave = threadIdx.x >> 5, lane = threadIdx.x & 31;
  int row0 = (blockIdx.x * 2 + wave) * 16;

  const float* arow =
      x + (size_t)(row0 + (lane & 15)) * IN_DIM + ((lane >> 4) << 3);

  v8f acc[4] = {};
  for (int k0 = 0; k0 < IN_DIM; k0 += 32) {
    if (k0 + 32 < IN_DIM) __builtin_prefetch(arow + k0 + 32, 0, 3);
    v4f f0 = *(const v4f*)(arow + k0);
    v4f f1 = *(const v4f*)(arow + k0 + 4);
    v4f f2 = *(const v4f*)(arow + k0 + 16);
    v4f f3 = *(const v4f*)(arow + k0 + 20);
    v16bf a;
#pragma unroll
    for (int i = 0; i < 4; ++i) {
      a[i]      = (__bf16)f0[i];
      a[4 + i]  = (__bf16)f1[i];
      a[8 + i]  = (__bf16)f2[i];
      a[12 + i] = (__bf16)f3[i];
    }
#pragma unroll
    for (int j = 0; j < 4; ++j) {
      v16bf b = load_b_packed(Wp, (k0 >> 5) * 4 + j, lane);
      acc[j]  = wmma_bf16(a, b, acc[j]);
    }
  }
  int rbase = (lane >> 4) * 8, col = lane & 15;
#pragma unroll
  for (int j = 0; j < 4; ++j) {
    float bv = bias[j * 16 + col];
#pragma unroll
    for (int r = 0; r < 8; ++r)
      tile[wave][rbase + r][j * 16 + col] = acc[j][r] + bv;
  }
  __syncthreads();
  if (lane < 16) {
    int rr = lane, n = row0 + rr;
    float mu = 0.f;
    for (int d = 0; d < HID; ++d) mu += tile[wave][rr][d];
    mu *= (1.0f / HID);
    float var = 0.f;
    for (int d = 0; d < HID; ++d) {
      float t = tile[wave][rr][d] - mu;
      var += t * t;
    }
    float is = rsqrtf(var * (1.0f / HID) + LN_EPS);
    for (int d = 0; d < HID; ++d) {
      float y = (tile[wave][rr][d] - mu) * is * lnw[d] + lnb[d];
      y = fmaxf(y, 0.0f);
      h[(size_t)n * HID + d]  = y;
      hb[(size_t)n * HID + d] = (__bf16)y;
    }
  }
}

// ---------------------------------------------------------------------------
// q/k/v = hb @ W + b (blockIdx.y selects which).
// q -> row-major qb[N][128];  k -> transposed kT[128][N] (vector b128 stores);
// v -> both vT[128][N] and row-major vb[N][128] (gcn gathers rows).
// Accumulates raw reductions Sq, Sk, ksum[c], vsum[c].
// ---------------------------------------------------------------------------
__global__ void k_qkv(const __bf16* __restrict__ hb,
                      const __bf16* __restrict__ wqp, const __bf16* __restrict__ wkp,
                      const __bf16* __restrict__ wvp,
                      const float* __restrict__ bq, const float* __restrict__ bk,
                      const float* __restrict__ bv,
                      __bf16* __restrict__ qb, __bf16* __restrict__ kT,
                      __bf16* __restrict__ vT, __bf16* __restrict__ vb,
                      float* Sq, float* Sk, float* ksum, float* vsum) {
  int which = blockIdx.y;
  const __bf16* W   = (which == 0) ? wqp : (which == 1) ? wkp : wvp;
  const float* bias = (which == 0) ? bq : (which == 1) ? bk : bv;

  int wave = threadIdx.x >> 5, lane = threadIdx.x & 31;
  int row0 = (blockIdx.x * 2 + wave) * 16;

  v8f acc[8] = {};
  for (int k0 = 0; k0 < HID; k0 += 32) {
    v16bf a = load_a_bf(hb, HID, row0, k0, lane);
#pragma unroll
    for (int j = 0; j < 8; ++j) {
      v16bf b = load_b_packed(W, (k0 >> 5) * 8 + j, lane);
      acc[j]  = wmma_bf16(a, b, acc[j]);
    }
  }
  int rbase = (lane >> 4) * 8, col = lane & 15;
  float ss = 0.f;
#pragma unroll
  for (int j = 0; j < 8; ++j) {
    int c = j * 16 + col;
    float bsv = bias[c];
    float colsum = 0.f;
    v8bf sv;
#pragma unroll
    for (int r = 0; r < 8; ++r) {
      float vvv = acc[j][r] + bsv;
      sv[r] = (__bf16)vvv;
      ss += vvv * vvv;
      colsum += vvv;
    }
    if (which == 0) {
#pragma unroll
      for (int r = 0; r < 8; ++r)
        qb[(size_t)(row0 + rbase + r) * 128 + c] = sv[r];
    } else if (which == 1) {
      *(v8bf*)(kT + (size_t)c * NN + row0 + rbase) = sv;
      unsafeAtomicAdd(&ksum[c], colsum);
    } else {
      *(v8bf*)(vT + (size_t)c * NN + row0 + rbase) = sv;
#pragma unroll
      for (int r = 0; r < 8; ++r)
        vb[(size_t)(row0 + rbase + r) * 128 + c] = sv[r];
      unsafeAtomicAdd(&vsum[c], colsum);
    }
  }
#pragma unroll
  for (int off = 16; off > 0; off >>= 1) ss += __shfl_xor(ss, off, 32);
  if (lane == 0) {
    if (which == 0) unsafeAtomicAdd(Sq, ss);
    else if (which == 1) unsafeAtomicAdd(Sk, ss);
  }
}

// ---------------------------------------------------------------------------
// kvs[h][m][d] += sum_n k[n,h,m]*v[n,h,d] via WMMA over node chunks, using the
// transposed kT/vT so every fragment load is a contiguous vector load.
// blockDim=32; grid=(chunks, 8); y = head*4 + m-tile.  NN % 32 == 0: no guards.
// ---------------------------------------------------------------------------
#define KVS_CHUNK 4096
__global__ void k_kvs(const __bf16* __restrict__ kT, const __bf16* __restrict__ vT,
                      float* __restrict__ kvs) {
  int lane = threadIdx.x & 31;
  int head = blockIdx.y >> 2;
  int mt   = blockIdx.y & 3;
  int node0 = blockIdx.x * KVS_CHUNK;
  int nodeEnd = node0 + KVS_CHUNK;
  if (nodeEnd > NN) nodeEnd = NN;

  const __bf16* arow = kT + (size_t)(head * 64 + mt * 16 + (lane & 15)) * NN +
                       ((lane >> 4) << 4);
  const __bf16* brow0 = vT + (size_t)(head * 64 + (lane & 15)) * NN +
                        ((lane >> 4) << 4);

  v8f acc[4] = {};
  for (int kk = node0; kk < nodeEnd; kk += 32) {
    v8bf lo = *(const v8bf*)(arow + kk);
    v8bf hi = *(const v8bf*)(arow + kk + 8);
    v16bf a;
#pragma unroll
    for (int i = 0; i < 8; ++i) { a[i] = lo[i]; a[8 + i] = hi[i]; }
#pragma unroll
    for (int j = 0; j < 4; ++j) {
      v16bf b = *(const v16bf*)(brow0 + (size_t)(j * 16) * NN + kk);
      acc[j]  = wmma_bf16(a, b, acc[j]);
    }
  }
  int mbase = mt * 16 + (lane >> 4) * 8;
  int d0 = lane & 15;
#pragma unroll
  for (int j = 0; j < 4; ++j)
#pragma unroll
    for (int r = 0; r < 8; ++r)
      unsafeAtomicAdd(&kvs[head * 4096 + (mbase + r) * 64 + j * 16 + d0],
                      acc[j][r]);
}

// ---------------------------------------------------------------------------
// Fold s = 1/(||q||*||k||) into kvs and pack it straight into per-lane WMMA
// B-fragment order ([head][ks*4+j][lane][16]); also ksum_s = ksum * s.
// ---------------------------------------------------------------------------
__global__ void k_scale(const float* __restrict__ red, const float* __restrict__ kvs,
                        float* __restrict__ ksum_s, __bf16* __restrict__ kvsb) {
  float s = rsqrtf(red[0]) * rsqrtf(red[1]);
  int t = blockIdx.x * blockDim.x + threadIdx.x;
  if (t < 128) ksum_s[t] = red[2 + t] * s;
  if (t < 8192) {
    int head = t >> 12;
    int u    = t & 4095;
    int i    = u & 15;
    int lane = (u >> 4) & 31;
    int frag = u >> 9;            // ks*4 + j
    int j = frag & 3, ks = frag >> 2;
    int k = ks * 32 + ((lane >> 4) << 3) + (i & 7) + ((i >> 3) << 4);
    int n = j * 16 + (lane & 15);
    kvsb[t] = (__bf16)(kvs[head * 4096 + k * 64 + n] * s);
  }
}

// ---------------------------------------------------------------------------
// GCN scatter: gcn[col] += w_e * v[row]; one wave per edge, 4 feats/lane.
// ---------------------------------------------------------------------------
__global__ void k_gcn(const int* __restrict__ row, const int* __restrict__ col,
                      const float* __restrict__ deg, const __bf16* __restrict__ vb,
                      float* __restrict__ gcn) {
  int wave = threadIdx.x >> 5, lane = threadIdx.x & 31;
  int e = blockIdx.x * 8 + wave;
  if (e >= EE) return;
  int r = row[e], c = col[e];
  float dr = deg[r], dc = deg[c];
  float w = (dr > 0.f && dc > 0.f) ? rsqrtf(dr) * rsqrtf(dc) : 0.f;
  if (w != 0.f) {
    v4bf vv = *(const v4bf*)(vb + (size_t)r * 128 + lane * 4);
    float* dst = gcn + (size_t)c * 128 + lane * 4;
#pragma unroll
    for (int i = 0; i < 4; ++i) unsafeAtomicAdd(&dst[i], w * (float)vv[i]);
  }
}

// ---------------------------------------------------------------------------
// attn num = q @ kvs_scaled (WMMA) + vsum; denom = q.ksum_s + N;
// out = mean_heads(num/denom + gcn); h = ALPHA*out + (1-ALPHA)*hin; LN.
// ---------------------------------------------------------------------------
__global__ void k_combine(const __bf16* __restrict__ qb, const __bf16* __restrict__ kvsb,
                          const float* __restrict__ ksum_s, const float* __restrict__ vsum,
                          const float* __restrict__ gcn, const float* __restrict__ hin,
                          const float* __restrict__ lnw, const float* __restrict__ lnb,
                          float* __restrict__ hout, __bf16* __restrict__ hbout) {
  __shared__ float num[2][16][136];
  __shared__ float den[2][16][2];
  int wave = threadIdx.x >> 5, lane = threadIdx.x & 31;
  int row0 = (blockIdx.x * 2 + wave) * 16;

  v8f acc[8] = {};
#pragma unroll
  for (int head = 0; head < 2; ++head)
    for (int k0 = 0; k0 < 64; k0 += 32) {
      v16bf a = load_a_bf(qb, 128, row0, head * 64 + k0, lane);
#pragma unroll
      for (int j = 0; j < 4; ++j) {
        v16bf b = load_b_packed(kvsb + head * 4096, (k0 >> 5) * 4 + j, lane);
        acc[head * 4 + j] = wmma_bf16(a, b, acc[head * 4 + j]);
      }
    }
  int rbase = (lane >> 4) * 8, col = lane & 15;
#pragma unroll
  for (int head = 0; head < 2; ++head)
#pragma unroll
    for (int j = 0; j < 4; ++j) {
      int c = head * 64 + j * 16 + col;
      float vs = vsum[c];
#pragma unroll
      for (int r = 0; r < 8; ++r)
        num[wave][rbase + r][c] = acc[head * 4 + j][r] + vs;
    }
  // denominators: lane t -> (row = t&15, head = t>>4)
  {
    int rr = lane & 15, hh = lane >> 4;
    int n = row0 + rr;
    const v8bf* qrow = (const v8bf*)(qb + (size_t)n * 128 + hh * 64);
    float dot = 0.f;
#pragma unroll
    for (int c8 = 0; c8 < 8; ++c8) {
      v8bf qs = qrow[c8];
#pragma unroll
      for (int i = 0; i < 8; ++i)
        dot += (float)qs[i] * ksum_s[hh * 64 + c8 * 8 + i];
    }
    den[wave][rr][hh] = dot + (float)NN;
  }
  __syncthreads();
  if (lane < 16) {
    int rr = lane, n = row0 + rr;
    float d0 = den[wave][rr][0], d1 = den[wave][rr][1];
    for (int d = 0; d < 64; ++d) {
      float o = 0.5f * ((num[wave][rr][d]      / d0 + gcn[(size_t)n * 128 + d]) +
                        (num[wave][rr][64 + d] / d1 + gcn[(size_t)n * 128 + 64 + d]));
      num[wave][rr][d] = ALPHA * o + (1.0f - ALPHA) * hin[(size_t)n * HID + d];
    }
    float mu = 0.f;
    for (int d = 0; d < 64; ++d) mu += num[wave][rr][d];
    mu *= (1.0f / 64.0f);
    float var = 0.f;
    for (int d = 0; d < 64; ++d) {
      float t = num[wave][rr][d] - mu;
      var += t * t;
    }
    float is = rsqrtf(var * (1.0f / 64.0f) + LN_EPS);
    for (int d = 0; d < 64; ++d) {
      float y = (num[wave][rr][d] - mu) * is * lnw[d] + lnb[d];
      hout[(size_t)n * HID + d]  = y;
      hbout[(size_t)n * HID + d] = (__bf16)y;
    }
  }
}

// ---------------------------------------------------------------------------
// Classifier: out = h @ fco_w + fco_b  (cols padded 40 -> 48, store guarded)
// ---------------------------------------------------------------------------
__global__ void k_out(const __bf16* __restrict__ hb, const __bf16* __restrict__ Wp,
                      const float* __restrict__ bias, float* __restrict__ out) {
  int wave = threadIdx.x >> 5, lane = threadIdx.x & 31;
  int row0 = (blockIdx.x * 2 + wave) * 16;
  v8f acc[3] = {};
  for (int k0 = 0; k0 < HID; k0 += 32) {
    v16bf a = load_a_bf(hb, HID, row0, k0, lane);
#pragma unroll
    for (int j = 0; j < 3; ++j) {
      v16bf b = load_b_packed(Wp, (k0 >> 5) * 3 + j, lane);
      acc[j]  = wmma_bf16(a, b, acc[j]);
    }
  }
  int rbase = (lane >> 4) * 8, c0 = lane & 15;
#pragma unroll
  for (int j = 0; j < 3; ++j) {
    int c = j * 16 + c0;
    if (c < 40) {
      float bv = bias[c];
#pragma unroll
      for (int r = 0; r < 8; ++r)
        out[(size_t)(row0 + rbase + r) * 40 + c] = acc[j][r] + bv;
    }
  }
}

// ---------------------------------------------------------------------------
extern "C" void kernel_launch(void* const* d_in, const int* in_sizes, int n_in,
                              void* d_out, int out_size, void* d_ws, size_t ws_size,
                              hipStream_t stream) {
  const float* x    = (const float*)d_in[0];
  const int*   ei   = (const int*)d_in[1];
  const int*   rowp = ei;
  const int*   colp = ei + EE;
  const float* fc0w = (const float*)d_in[2];
  const float* fc0b = (const float*)d_in[3];
  const float* lnw  = (const float*)d_in[4];
  const float* lnb  = (const float*)d_in[5];
  const float* Wq   = (const float*)d_in[6];
  const float* Wk   = (const float*)d_in[7];
  const float* Wv   = (const float*)d_in[8];
  const float* bq   = (const float*)d_in[9];
  const float* bk   = (const float*)d_in[10];
  const float* bv   = (const float*)d_in[11];
  const float* fcow = (const float*)d_in[12];
  const float* fcob = (const float*)d_in[13];

  // ---- carve workspace (~220 MB) ----
  uintptr_t base = (uintptr_t)d_ws;
  auto carve = [&](size_t bytes) -> void* {
    void* r = (void*)base;
    base += (bytes + 255) & ~(size_t)255;
    return r;
  };
  float*  h0    = (float*) carve((size_t)NN * HID * 4);
  float*  h1    = (float*) carve((size_t)NN * HID * 4);
  __bf16* hb    = (__bf16*)carve((size_t)NN * HID * 2);
  __bf16* qb    = (__bf16*)carve((size_t)NN * 128 * 2);
  __bf16* kT    = (__bf16*)carve((size_t)NN * 128 * 2);
  __bf16* vT    = (__bf16*)carve((size_t)NN * 128 * 2);
  __bf16* vb    = (__bf16*)carve((size_t)NN * 128 * 2);
  float*  gcn   = (float*) carve((size_t)NN * 128 * 4);
  float*  deg   = (float*) carve((size_t)NN * 4);
  __bf16* wfc0p = (__bf16*)carve((size_t)IN_DIM * HID * 2);
  __bf16* wqp   = (__bf16*)carve((size_t)NLAYERS * HID * 128 * 2);
  __bf16* wkp   = (__bf16*)carve((size_t)NLAYERS * HID * 128 * 2);
  __bf16* wvp   = (__bf16*)carve((size_t)NLAYERS * HID * 128 * 2);
  __bf16* wobp  = (__bf16*)carve((size_t)HID * 48 * 2);
  float*  red   = (float*) carve((size_t)(386 + 8192) * 4);
  __bf16* kvsb  = (__bf16*)carve((size_t)8192 * 2);

  float* Sq     = red + 0;
  float* Sk     = red + 1;
  float* ksum   = red + 2;
  float* vsum   = red + 130;
  float* ksum_s = red + 258;
  float* kvs    = red + 386;

  // ---- weight packing (fp32 -> bf16, WMMA fragment order) ----
  k_packB<<<dim3((IN_DIM * HID + 255) / 256), 256, 0, stream>>>(
      fc0w, wfc0p, IN_DIM, HID, HID);
  for (int l = 0; l < NLAYERS; ++l) {
    k_packB<<<dim3((HID * 128 + 255) / 256), 256, 0, stream>>>(
        Wq + l * HID * 128, wqp + l * HID * 128, HID, 128, 128);
    k_packB<<<dim3((HID * 128 + 255) / 256), 256, 0, stream>>>(
        Wk + l * HID * 128, wkp + l * HID * 128, HID, 128, 128);
    k_packB<<<dim3((HID * 128 + 255) / 256), 256, 0, stream>>>(
        Wv + l * HID * 128, wvp + l * HID * 128, HID, 128, 128);
  }
  k_packB<<<dim3((HID * 48 + 255) / 256), 256, 0, stream>>>(
      fcow, wobp, HID, 48, 40);

  // ---- input projection + LN + relu ----
  k_fc0<<<dim3(NN / 32), 64, 0, stream>>>(x, wfc0p, fc0b, lnw, lnb, h0, hb);

  // ---- degrees (shared by both layers) ----
  k_zero<<<dim3((NN + 255) / 256), 256, 0, stream>>>(deg, NN);
  k_deg<<<dim3((EE + 255) / 256), 256, 0, stream>>>(colp, deg);

  float* hin = h0;
  float* hout = h1;
  for (int l = 0; l < NLAYERS; ++l) {
    k_zero<<<dim3((386 + 8192 + 255) / 256), 256, 0, stream>>>(red, 386 + 8192);
    k_zero<<<dim3((NN * 128 + 255) / 256), 256, 0, stream>>>(gcn, NN * 128);

    k_qkv<<<dim3(NN / 32, 3), 64, 0, stream>>>(
        hb, wqp + l * HID * 128, wkp + l * HID * 128, wvp + l * HID * 128,
        bq + l * 128, bk + l * 128, bv + l * 128,
        qb, kT, vT, vb, Sq, Sk, ksum, vsum);

    k_kvs<<<dim3((NN + KVS_CHUNK - 1) / KVS_CHUNK, 8), 32, 0, stream>>>(kT, vT, kvs);
    k_scale<<<dim3(32), 256, 0, stream>>>(red, kvs, ksum_s, kvsb);
    k_gcn<<<dim3(EE / 8), 256, 0, stream>>>(rowp, colp, deg, vb, gcn);

    k_combine<<<dim3(NN / 32), 64, 0, stream>>>(
        qb, kvsb, ksum_s, vsum, gcn, hin,
        lnw + (l + 1) * HID, lnb + (l + 1) * HID, hout, hb);

    float* t = hin; hin = hout; hout = t;
  }

  k_out<<<dim3(NN / 32), 64, 0, stream>>>(hb, wobp, fcob, (float*)d_out);
}
